// NetConv_57939108823648
// MI455X (gfx1250) — compile-verified, matrix operand order checked by reading, not run on hardware
//
#include <hip/hip_runtime.h>
#include <math.h>

// ---------------------------------------------------------------------------
// CDNA5 (gfx1250, wave32) WMMA implementation of the NetConv GNN layer.
//
// Edge MLPs (40 -> 16 -> 16/17) mapped to V_WMMA_F32_16X16X32_F16:
//   * 16 edges per wave form the M=16 dimension.
//   * K=40 input padded to 64 -> 2 WMMAs for layer 1; K=16 hidden padded to
//     32 -> 1 WMMA for layer 2. f32 accumulation, bias preloaded into C.
//   * LeakyReLU(0.2) == max(x, 0.2x) on the C fragment in VALU.
//   * C fragment (M = vgpr + 8*(lane>>4), N = lane&15) scattered with
//     global f32 atomics into L2-resident node accumulators.
// Node reduce MLP (32 -> 16 -> 16): one K=32 WMMA + one K=32-padded WMMA.
// LDS staging uses packed v4h (ds_store_b64) writes and b128 fragment reads.
// ---------------------------------------------------------------------------

typedef __attribute__((ext_vector_type(16))) _Float16 v16h;
typedef __attribute__((ext_vector_type(4)))  _Float16 v4h;
typedef __attribute__((ext_vector_type(8)))  float    v8f;

#define NNODES 100000
#define NEDGES 2000000

// 16-bit A-matrix 16x32 element->K mapping (ISA 7.12.2):
// lane half h=lane>>4; vector element e in 0..15 holds K = (e&7) + 8h + 16*(e>>3)
__device__ __forceinline__ int kmap(int e, int h) {
  return (e & 7) + 8 * h + ((e >> 3) << 4);
}

__device__ __forceinline__ v4h pack4(float4 v) {
  v4h r;
  r[0] = (_Float16)v.x; r[1] = (_Float16)v.y;
  r[2] = (_Float16)v.z; r[3] = (_Float16)v.w;
  return r;
}

// Build a B fragment (K x 16, K in [kbase, kbase+32)) for column n from a
// row-major f32 weight matrix w[K][stride]; rows >= kmax are zero padding.
__device__ __forceinline__ v16h load_b_frag(const float* __restrict__ w, int stride,
                                            int n, int kbase, int kmax, int h) {
  v16h b;
#pragma unroll
  for (int e = 0; e < 16; ++e) {
    int K = kbase + kmap(e, h);
    b[e] = (K < kmax) ? (_Float16)w[K * stride + n] : (_Float16)0.0f;
  }
  return b;
}

// Load an A fragment for this lane's row (already staged in LDS, f16,
// row pointer given, zero-padded in memory up to kbase+32).
__device__ __forceinline__ v16h load_a_frag(const _Float16* row, int kbase, int h) {
  v16h a;
#pragma unroll
  for (int e = 0; e < 16; ++e) a[e] = row[kbase + kmap(e, h)];
  return a;
}

// A fragment for the K=16 hidden layer (K >= 16 is zero, not in memory).
__device__ __forceinline__ v16h load_a_frag_hid(const _Float16* row, int h) {
  v16h a;
#pragma unroll
  for (int e = 0; e < 16; ++e) {
    int K = kmap(e, h);
    a[e] = (K < 16) ? row[K] : (_Float16)0.0f;
  }
  return a;
}

__device__ __forceinline__ void lds_fence() {
  asm volatile("s_wait_dscnt 0x0" ::: "memory");
}

// Gather 16 f32 node features -> 4 packed ds_store_b64 at stage[off..off+15].
__device__ __forceinline__ void stage_row16(_Float16* dstp, const float* srcp) {
  const float4* p = (const float4*)srcp;
#pragma unroll
  for (int j = 0; j < 4; ++j) *(v4h*)(dstp + 4 * j) = pack4(p[j]);
}

__device__ __forceinline__ void stage_row8(_Float16* dstp, const float* srcp) {
  const float4* p = (const float4*)srcp;
#pragma unroll
  for (int j = 0; j < 2; ++j) *(v4h*)(dstp + 4 * j) = pack4(p[j]);
}

__global__ void zero_kernel(float* __restrict__ p, long n) {
  long i = (long)blockIdx.x * blockDim.x + threadIdx.x;
  long stride = (long)gridDim.x * blockDim.x;
  for (; i < n; i += stride) p[i] = 0.0f;
}

// --------------------------- edge MLP: o2i ---------------------------------
__global__ void __launch_bounds__(256)
edge_o2i_kernel(const float* __restrict__ nf,
                const int* __restrict__ src, const int* __restrict__ dst,
                const float* __restrict__ nef,
                const float* __restrict__ w1, const float* __restrict__ b1,
                const float* __restrict__ w2, const float* __restrict__ b2,
                float* __restrict__ acc /* [NNODES,16] */) {
  __shared__ _Float16 stage_s[8][16 * 64];
  __shared__ _Float16 hid_s[8][16 * 16];
  const int lane = threadIdx.x & 31;
  const int wid  = threadIdx.x >> 5;
  const int m = lane & 15;   // A row / B-C column for this lane
  const int h = lane >> 4;   // K half select
  _Float16* stage = stage_s[wid];
  _Float16* hid   = hid_s[wid];

  // Weight fragments: built once per wave (tiny, L2 resident).
  const v16h b1a = load_b_frag(w1, 16, m, 0, 40, h);
  const v16h b1b = load_b_frag(w1, 16, m, 32, 40, h);
  const v16h b2f = load_b_frag(w2, 16, m, 0, 16, h);
  const float bias1 = b1[m];
  const float bias2 = b2[m];

  // Zero pad K=40..63 once (constant across tiles).
  if (h == 1) {
    const v4h z = {};
#pragma unroll
    for (int k = 40; k < 64; k += 4) *(v4h*)(stage + m * 64 + k) = z;
  }

  const int nTiles = NEDGES / 16;
  const int wavesTotal = (int)((gridDim.x * blockDim.x) >> 5);
  for (int tile = (int)(blockIdx.x * (blockDim.x >> 5)) + wid; tile < nTiles;
       tile += wavesTotal) {
    const int base = tile * 16;
    // Gather concat[nf[src], nf[dst], nef] -> 16x64 f16 tile in LDS.
    {
      const int e = base + m;
      if (h == 0) {
        stage_row16(stage + m * 64, nf + (long)src[e] * 16);
      } else {
        stage_row16(stage + m * 64 + 16, nf + (long)dst[e] * 16);
        stage_row8(stage + m * 64 + 32, nef + (long)e * 8);
      }
    }
    lds_fence();

    // Layer 1: [16x40] @ [40x16] (K padded to 64) -> hidden C, +bias, LeakyReLU.
    v16h a0 = load_a_frag(stage + m * 64, 0, h);
    v16h a1 = load_a_frag(stage + m * 64, 32, h);
    v8f c;
#pragma unroll
    for (int i = 0; i < 8; ++i) c[i] = bias1;
    c = __builtin_amdgcn_wmma_f32_16x16x32_f16(false, a0, false, b1a, (short)0, c, false, false);
    c = __builtin_amdgcn_wmma_f32_16x16x32_f16(false, a1, false, b1b, (short)0, c, false, false);
#pragma unroll
    for (int i = 0; i < 8; ++i) c[i] = fmaxf(c[i], 0.2f * c[i]);

    // Bounce C (row-striped-across-lanes) to A layout through LDS.
#pragma unroll
    for (int i = 0; i < 8; ++i) hid[(i + 8 * h) * 16 + m] = (_Float16)c[i];
    lds_fence();

    // Layer 2: [16x16] @ [16x16] (K padded to 32) + bias.
    v16h a2 = load_a_frag_hid(hid + m * 16, h);
    v8f o;
#pragma unroll
    for (int i = 0; i < 8; ++i) o[i] = bias2;
    o = __builtin_amdgcn_wmma_f32_16x16x32_f16(false, a2, false, b2f, (short)0, o, false, false);

    // Scatter-sum into destination nodes. This lane's 8 rows are contiguous:
    // M = i + 8h -> dst[base+8h .. base+8h+7], load as two int4.
    const int4 d0 = *(const int4*)(dst + base + 8 * h);
    const int4 d1 = *(const int4*)(dst + base + 8 * h + 4);
    const int didx[8] = {d0.x, d0.y, d0.z, d0.w, d1.x, d1.y, d1.z, d1.w};
#pragma unroll
    for (int i = 0; i < 8; ++i)
      atomicAdd(&acc[(long)didx[i] * 16 + m], o[i]);
  }
}

// --------------------------- edge MLP: i2o (gated) -------------------------
__global__ void __launch_bounds__(256)
edge_i2o_kernel(const float* __restrict__ nf,
                const int* __restrict__ src, const int* __restrict__ dst,
                const float* __restrict__ nef,
                const float* __restrict__ w1, const float* __restrict__ b1,
                const float* __restrict__ w2 /* [16][17] */,
                const float* __restrict__ b2 /* [17] */,
                float* __restrict__ nfo1 /* [NNODES,8] */,
                float* __restrict__ nfo2 /* [NNODES,8] */,
                float* __restrict__ cnt  /* [NNODES] */) {
  __shared__ _Float16 stage_s[8][16 * 64];
  __shared__ _Float16 hid_s[8][16 * 16];
  __shared__ float    gate_s[8][16];
  const int lane = threadIdx.x & 31;
  const int wid  = threadIdx.x >> 5;
  const int m = lane & 15;
  const int h = lane >> 4;
  _Float16* stage = stage_s[wid];
  _Float16* hid   = hid_s[wid];
  float*    gate  = gate_s[wid];

  const v16h b1a = load_b_frag(w1, 16, m, 0, 40, h);
  const v16h b1b = load_b_frag(w1, 16, m, 32, 40, h);
  // Layer-2 WMMA covers output columns 1..16 of the 17-wide MLP output.
  v16h b2f;
#pragma unroll
  for (int e = 0; e < 16; ++e) {
    int K = kmap(e, h);
    b2f[e] = (K < 16) ? (_Float16)w2[K * 17 + (m + 1)] : (_Float16)0.0f;
  }
  const float bias1 = b1[m];
  const float bias2 = b2[m + 1];
  // Gate column (output 0) weights, kept in registers.
  float gw[16];
#pragma unroll
  for (int k = 0; k < 16; ++k) gw[k] = w2[k * 17];
  const float gb = b2[0];

  if (h == 1) {
    const v4h z = {};
#pragma unroll
    for (int k = 40; k < 64; k += 4) *(v4h*)(stage + m * 64 + k) = z;
  }

  const int nTiles = NEDGES / 16;
  const int wavesTotal = (int)((gridDim.x * blockDim.x) >> 5);
  for (int tile = (int)(blockIdx.x * (blockDim.x >> 5)) + wid; tile < nTiles;
       tile += wavesTotal) {
    const int base = tile * 16;
    {
      const int e = base + m;
      if (h == 0) {
        stage_row16(stage + m * 64, nf + (long)src[e] * 16);
      } else {
        stage_row16(stage + m * 64 + 16, nf + (long)dst[e] * 16);
        stage_row8(stage + m * 64 + 32, nef + (long)e * 8);
      }
    }
    lds_fence();

    v16h a0 = load_a_frag(stage + m * 64, 0, h);
    v16h a1 = load_a_frag(stage + m * 64, 32, h);
    v8f c;
#pragma unroll
    for (int i = 0; i < 8; ++i) c[i] = bias1;
    c = __builtin_amdgcn_wmma_f32_16x16x32_f16(false, a0, false, b1a, (short)0, c, false, false);
    c = __builtin_amdgcn_wmma_f32_16x16x32_f16(false, a1, false, b1b, (short)0, c, false, false);
#pragma unroll
    for (int i = 0; i < 8; ++i) c[i] = fmaxf(c[i], 0.2f * c[i]);

#pragma unroll
    for (int i = 0; i < 8; ++i) hid[(i + 8 * h) * 16 + m] = (_Float16)c[i];
    lds_fence();

    // Gate: sigmoid of hidden . w2[:,0] + b2[0], one row per lane (lanes 0..15).
    if (h == 0) {
      float x = gb;
#pragma unroll
      for (int k = 0; k < 16; ++k) x += (float)hid[m * 16 + k] * gw[k];
      gate[m] = 1.0f / (1.0f + __expf(-x));
    }

    v16h a2 = load_a_frag_hid(hid + m * 16, h);
    v8f o;
#pragma unroll
    for (int i = 0; i < 8; ++i) o[i] = bias2;
    o = __builtin_amdgcn_wmma_f32_16x16x32_f16(false, a2, false, b2f, (short)0, o, false, false);
    lds_fence();  // gate[] visible

    // Gated scatter: cols 0..7 -> nfo1, cols 8..15 -> nfo2; +1 per edge to cnt.
    const int4 d0 = *(const int4*)(dst + base + 8 * h);
    const int4 d1 = *(const int4*)(dst + base + 8 * h + 4);
    const int didx[8] = {d0.x, d0.y, d0.z, d0.w, d1.x, d1.y, d1.z, d1.w};
    float* outcol = (m < 8) ? nfo1 : nfo2;
    const int col = m & 7;
#pragma unroll
    for (int i = 0; i < 8; ++i) {
      float v = o[i] * gate[i + 8 * h];
      atomicAdd(&outcol[(long)didx[i] * 8 + col], v);
    }
    if (lane < 16) atomicAdd(&cnt[dst[base + lane]], 1.0f);
  }
}

// --------------------------- node reduce MLP -------------------------------
__global__ void __launch_bounds__(256)
node_reduce_kernel(const float* __restrict__ acc,   // new_nf [NNODES,16]
                   const float* __restrict__ nfo1,  // [NNODES,8]
                   const float* __restrict__ nfo2,  // [NNODES,8]
                   const float* __restrict__ cnt,   // [NNODES]
                   const float* __restrict__ w1, const float* __restrict__ b1,
                   const float* __restrict__ w2, const float* __restrict__ b2,
                   float* __restrict__ out /* [NNODES,16] */) {
  __shared__ _Float16 stage_s[8][16 * 32];
  __shared__ _Float16 hid_s[8][16 * 16];
  const int lane = threadIdx.x & 31;
  const int wid  = threadIdx.x >> 5;
  const int m = lane & 15;
  const int h = lane >> 4;
  _Float16* stage = stage_s[wid];
  _Float16* hid   = hid_s[wid];

  const v16h bredA = load_b_frag(w1, 16, m, 0, 32, h);  // K=32 exact
  const v16h bredB = load_b_frag(w2, 16, m, 0, 16, h);  // K=16 padded to 32
  const float bias1 = b1[m];
  const float bias2 = b2[m];

  const int nTiles = NNODES / 16;  // 6250 exactly
  const int wavesTotal = (int)((gridDim.x * blockDim.x) >> 5);
  for (int tile = (int)(blockIdx.x * (blockDim.x >> 5)) + wid; tile < nTiles;
       tile += wavesTotal) {
    const int base = tile * 16;
    {
      const int node = base + m;
      if (h == 0) {
        stage_row16(stage + m * 32, acc + (long)node * 16);
      } else {
        const float inv = 1.0f / fmaxf(cnt[node], 1.0f);
        stage_row8(stage + m * 32 + 16, nfo1 + (long)node * 8);
        const float4* q = (const float4*)(nfo2 + (long)node * 8);
#pragma unroll
        for (int j = 0; j < 2; ++j) {
          float4 v = q[j];
          v.x *= inv; v.y *= inv; v.z *= inv; v.w *= inv;
          *(v4h*)(stage + m * 32 + 24 + 4 * j) = pack4(v);
        }
      }
    }
    lds_fence();

    v16h a0 = load_a_frag(stage + m * 32, 0, h);  // K=32: [new_nf|nfo1|nfo2/cnt]
    v8f c;
#pragma unroll
    for (int i = 0; i < 8; ++i) c[i] = bias1;
    c = __builtin_amdgcn_wmma_f32_16x16x32_f16(false, a0, false, bredA, (short)0, c, false, false);
#pragma unroll
    for (int i = 0; i < 8; ++i) c[i] = fmaxf(c[i], 0.2f * c[i]);

#pragma unroll
    for (int i = 0; i < 8; ++i) hid[(i + 8 * h) * 16 + m] = (_Float16)c[i];
    lds_fence();

    v16h a2 = load_a_frag_hid(hid + m * 16, h);
    v8f o;
#pragma unroll
    for (int i = 0; i < 8; ++i) o[i] = bias2;
    o = __builtin_amdgcn_wmma_f32_16x16x32_f16(false, a2, false, bredB, (short)0, o, false, false);

    // Select per node: reduced if it had in-edges, else new_nf.
    // This lane's rows are nodes base+8h .. base+8h+7 -> contiguous cnt reads.
    const float4 c0 = *(const float4*)(cnt + base + 8 * h);
    const float4 c1 = *(const float4*)(cnt + base + 8 * h + 4);
    const float cv[8] = {c0.x, c0.y, c0.z, c0.w, c1.x, c1.y, c1.z, c1.w};
#pragma unroll
    for (int i = 0; i < 8; ++i) {
      int node = base + i + 8 * h;
      float val = (cv[i] > 0.0f) ? o[i] : acc[(long)node * 16 + m];
      out[(long)node * 16 + m] = val;
    }
  }
}

// ---------------------------------------------------------------------------
extern "C" void kernel_launch(void* const* d_in, const int* in_sizes, int n_in,
                              void* d_out, int out_size, void* d_ws, size_t ws_size,
                              hipStream_t stream) {
  const float* nf      = (const float*)d_in[0];
  const int*   src_out = (const int*)d_in[1];
  const int*   dst_out = (const int*)d_in[2];
  const float* nef_out = (const float*)d_in[3];
  const int*   src_in  = (const int*)d_in[4];
  const int*   dst_in  = (const int*)d_in[5];
  const float* nef_in  = (const float*)d_in[6];
  const float* w_o2i1  = (const float*)d_in[7];
  const float* b_o2i1  = (const float*)d_in[8];
  const float* w_o2i2  = (const float*)d_in[9];
  const float* b_o2i2  = (const float*)d_in[10];
  const float* w_i2o1  = (const float*)d_in[11];
  const float* b_i2o1  = (const float*)d_in[12];
  const float* w_i2o2  = (const float*)d_in[13];
  const float* b_i2o2  = (const float*)d_in[14];
  const float* w_red1  = (const float*)d_in[15];
  const float* b_red1  = (const float*)d_in[16];
  const float* w_red2  = (const float*)d_in[17];
  const float* b_red2  = (const float*)d_in[18];
  float* out = (float*)d_out;

  // Workspace layout (f32): new_nf[N,16] | nfo1[N,8] | nfo2[N,8] | cnt[N]
  float* acc  = (float*)d_ws;
  float* nfo1 = acc + (long)NNODES * 16;
  float* nfo2 = nfo1 + (long)NNODES * 8;
  float* cnt  = nfo2 + (long)NNODES * 8;
  const long ws_floats = (long)NNODES * 33;

  zero_kernel<<<256, 256, 0, stream>>>(acc, ws_floats);
  edge_o2i_kernel<<<1024, 256, 0, stream>>>(nf, src_out, dst_out, nef_out,
                                            w_o2i1, b_o2i1, w_o2i2, b_o2i2, acc);
  edge_i2o_kernel<<<1024, 256, 0, stream>>>(nf, src_in, dst_in, nef_in,
                                            w_i2o1, b_i2o1, w_i2o2, b_i2o2,
                                            nfo1, nfo2, cnt);
  node_reduce_kernel<<<128, 256, 0, stream>>>(acc, nfo1, nfo2, cnt,
                                              w_red1, b_red1, w_red2, b_red2, out);
  (void)in_sizes; (void)n_in; (void)out_size; (void)ws_size;
}
